// AttentionControl_8220567404655
// MI455X (gfx1250) — compile-verified, matrix operand order checked by reading, not run on hardware
//
#include <hip/hip_runtime.h>
#include <hip/hip_bf16.h>

typedef __attribute__((ext_vector_type(16))) _Float16 v16h;
typedef __attribute__((ext_vector_type(8)))  float    v8f;
typedef __attribute__((ext_vector_type(8)))  unsigned int v8u;

union Frag {
    v16h h;
    v8u  u;
    unsigned int w[8];
};

#define IMGS     224
#define BQ       16          // batch rows per block in main kernel
#define GSTRIDE  776         // 768 + 8 pad (halfs), even -> u32-aligned pair reads
#define S2STRIDE 264         // 256 + 8 pad (halfs)
#define N1PACK   (24*8*32*16)   // W_hg packed halfs  (K=768 -> 24 ksteps, 8 ntiles)
#define N2PACK   (8*16*32*16)   // Wc  packed halfs   (K=256 -> 8 ksteps, 16 ntiles)

__device__ __forceinline__ float samp(const float* __restrict__ im, int y, int x) {
    return ((unsigned)y < (unsigned)IMGS && (unsigned)x < (unsigned)IMGS)
               ? im[y * IMGS + x] : 0.0f;
}

// ---------------------------------------------------------------------------
// Kernel 1: pre-pack W_hg[768,128] and Wc=[W_gs;W_ls][256,256] into f16
// WMMA B-fragment order: [kstep][ntile][lane(32)][16 halfs]. One contiguous
// 32B chunk per lane per fragment in the main kernel.
// 16-bit operand layout (ISA 7.12.2): lane<16 covers k-offsets {0..7,16..23},
// lane>=16 covers {8..15,24..31}; half j<4 -> kb+2j+p, j>=4 -> kb+16+2(j-4)+p.
// ---------------------------------------------------------------------------
__global__ __launch_bounds__(256) void attn_pack_weights(
    const float* __restrict__ W_hg,   // [768,128]
    const float* __restrict__ W_gs,   // [128,256]
    const float* __restrict__ W_ls,   // [128,256]
    _Float16* __restrict__ packWhg,   // [N1PACK]
    _Float16* __restrict__ packWc)    // [N2PACK]
{
    int tid = blockIdx.x * 256 + threadIdx.x;
    if (tid < N1PACK) {
        int p = tid & 15, j = p >> 1, pr = p & 1;
        int lane = (tid >> 4) & 31;
        int nt   = (tid >> 9) & 7;
        int ks   = tid >> 12;
        int kb   = (lane < 16) ? 0 : 8;
        int klocal = kb + ((j < 4) ? (2 * j + pr) : (16 + 2 * (j - 4) + pr));
        int k = ks * 32 + klocal;
        int n = nt * 16 + (lane & 15);
        packWhg[tid] = (_Float16)W_hg[k * 128 + n];
    } else if (tid < N1PACK + N2PACK) {
        int t = tid - N1PACK;
        int p = t & 15, j = p >> 1, pr = p & 1;
        int lane = (t >> 4) & 31;
        int nt   = (t >> 9) & 15;
        int ks   = t >> 13;
        int kb   = (lane < 16) ? 0 : 8;
        int klocal = kb + ((j < 4) ? (2 * j + pr) : (16 + 2 * (j - 4) + pr));
        int k = ks * 32 + klocal;
        int n = nt * 16 + (lane & 15);
        float v = (k < 128) ? W_gs[k * 256 + n] : W_ls[(k - 128) * 256 + n];
        packWc[t] = (_Float16)v;
    }
}

// ---------------------------------------------------------------------------
// Kernel 2: loc = hard_tanh(output @ W_loc + b_loc); one wave32 per row.
// loc_int = rint((loc+1)/2 * 224)  (RNE, matches jnp.round).
// ---------------------------------------------------------------------------
__global__ __launch_bounds__(256) void attn_loc(
    const float* __restrict__ state,   // [512,512]
    const float* __restrict__ W_loc,   // [512,2]
    const float* __restrict__ b_loc,   // [2]
    float* __restrict__ locf,          // [512,2]
    int*   __restrict__ loci)          // [512,2]
{
    int row  = (blockIdx.x * blockDim.x + threadIdx.x) >> 5;
    int lane = threadIdx.x & 31;
    if (row >= 512) return;
    float a0 = 0.0f, a1 = 0.0f;
    const float* s = state + (size_t)row * 512;
    for (int k = lane; k < 512; k += 32) {
        float v = s[k];
        a0 += v * W_loc[k * 2 + 0];
        a1 += v * W_loc[k * 2 + 1];
    }
    for (int off = 16; off; off >>= 1) {
        a0 += __shfl_xor(a0, off);
        a1 += __shfl_xor(a1, off);
    }
    if (lane == 0) {
        float l0 = fminf(fmaxf(a0 + b_loc[0], -1.0f), 1.0f);
        float l1 = fminf(fmaxf(a1 + b_loc[1], -1.0f), 1.0f);
        locf[row * 2 + 0] = l0;
        locf[row * 2 + 1] = l1;
        loci[row * 2 + 0] = (int)rintf((l0 + 1.0f) * 0.5f * (float)IMGS);
        loci[row * 2 + 1] = (int)rintf((l1 + 1.0f) * 0.5f * (float)IMGS);
    }
}

// ---------------------------------------------------------------------------
// Kernel 3: fused glimpse gather + both WMMA GEMMs. 16 batch rows per block,
// 8 waves. All intermediates stay in LDS; output written once.
// ---------------------------------------------------------------------------
__global__ __launch_bounds__(256) void attn_main(
    const float* __restrict__ img,      // [512,224,224]
    const float* __restrict__ W_hl,     // [2,128]
    const float* __restrict__ b_hl,     // [128]
    const float* __restrict__ b_hg,     // [128]
    const float* __restrict__ b_gs,     // [256]
    const float* __restrict__ b_ls,     // [256]
    const float* __restrict__ locf,     // [512,2]
    const int*   __restrict__ loci,     // [512,2]
    const _Float16* __restrict__ packWhg,
    const _Float16* __restrict__ packWc,
    float* __restrict__ out)            // [512,256]
{
    __shared__ __align__(16) _Float16 shG[BQ * GSTRIDE];    // glimpses f16 (stage-1 A)
    __shared__ __align__(16) _Float16 shA2[BQ * S2STRIDE];  // [hg | hl] f16 (stage-2 A)
    __shared__ int shLoc[BQ * 2];

    const int tid = threadIdx.x;
    const int r0  = blockIdx.x * BQ;

    if (tid < BQ * 2) shLoc[tid] = loci[r0 * 2 + tid];
    __syncthreads();

    // ---- phase 1a: multi-scale glimpse gather -> shG (f16) ----------------
    for (int t = tid; t < BQ * 768; t += 256) {
        int m = t / 768, k = t - m * 768;
        int depth = k >> 8, idx = k & 255;
        int gy = idx >> 4, gx = idx & 15;
        int lr = shLoc[m * 2 + 0], lc = shLoc[m * 2 + 1];
        const float* im = img + (size_t)(r0 + m) * IMGS * IMGS;
        float v;
        if (depth == 0) {
            v = samp(im, lr - 8 + gy, lc - 8 + gx);
        } else if (depth == 1) {           // 32x32 -> 16x16, centers at 2g+0.5
            int y = lr - 16 + 2 * gy, x = lc - 16 + 2 * gx;
            v = 0.25f * (samp(im, y, x) + samp(im, y, x + 1) +
                         samp(im, y + 1, x) + samp(im, y + 1, x + 1));
        } else {                           // 64x64 -> 16x16, centers at 4g+1.5
            int y = lr - 32 + 4 * gy + 1, x = lc - 32 + 4 * gx + 1;
            v = 0.25f * (samp(im, y, x) + samp(im, y, x + 1) +
                         samp(im, y + 1, x) + samp(im, y + 1, x + 1));
        }
        shG[m * GSTRIDE + k] = (_Float16)v;
    }

    // ---- phase 1b: hl = relu(loc @ W_hl + b_hl) -> shA2[:,128:256] --------
    #pragma unroll
    for (int i = 0; i < 8; ++i) {
        int idx = tid * 8 + i;             // 0..2047
        int m = idx >> 7, n = idx & 127;
        float l0 = locf[(r0 + m) * 2 + 0];
        float l1 = locf[(r0 + m) * 2 + 1];
        float v  = l0 * W_hl[n] + l1 * W_hl[128 + n] + b_hl[n];
        shA2[m * S2STRIDE + 128 + n] = (_Float16)fmaxf(v, 0.0f);
    }
    __syncthreads();

    const int lane  = tid & 31;
    const int wave  = tid >> 5;
    const int nloc  = lane & 15;
    const int mA    = lane & 15;
    const int kbase = (lane < 16) ? 0 : 8;
    const int mrow  = (lane < 16) ? 0 : 8;

    // ---- stage 1: hg[16,128] = relu(G @ W_hg + b_hg), wave -> n-tile ------
    {
        v8f acc = {};
        for (int ks = 0; ks < 24; ++ks) {
            Frag a, b;
            #pragma unroll
            for (int j = 0; j < 8; ++j) {
                int klo = ks * 32 + kbase + ((j < 4) ? (2 * j) : (16 + 2 * (j - 4)));
                a.w[j] = *reinterpret_cast<const unsigned int*>(&shG[mA * GSTRIDE + klo]);
            }
            b.u = *reinterpret_cast<const v8u*>(packWhg + (((ks * 8 + wave) * 32 + lane) << 4));
            acc = __builtin_amdgcn_wmma_f32_16x16x32_f16(
                false, a.h, false, b.h, (short)0, acc, false, false);
        }
        int n = wave * 16 + nloc;
        float bias = b_hg[n];
        #pragma unroll
        for (int r = 0; r < 8; ++r) {
            float v = fmaxf(acc[r] + bias, 0.0f);
            shA2[(mrow + r) * S2STRIDE + n] = (_Float16)v;
        }
    }
    __syncthreads();

    // ---- stage 2: g[16,256] = relu([hg|hl] @ [Wgs;Wls] + b_gs + b_ls) -----
    #pragma unroll
    for (int tt = 0; tt < 2; ++tt) {
        int nt = wave + tt * 8;
        v8f acc = {};
        for (int ks = 0; ks < 8; ++ks) {
            Frag a, b;
            #pragma unroll
            for (int j = 0; j < 8; ++j) {
                int klo = ks * 32 + kbase + ((j < 4) ? (2 * j) : (16 + 2 * (j - 4)));
                a.w[j] = *reinterpret_cast<const unsigned int*>(&shA2[mA * S2STRIDE + klo]);
            }
            b.u = *reinterpret_cast<const v8u*>(packWc + (((ks * 16 + nt) * 32 + lane) << 4));
            acc = __builtin_amdgcn_wmma_f32_16x16x32_f16(
                false, a.h, false, b.h, (short)0, acc, false, false);
        }
        int n = nt * 16 + nloc;
        float bias = b_gs[n] + b_ls[n];
        #pragma unroll
        for (int r = 0; r < 8; ++r) {
            out[(size_t)(r0 + mrow + r) * 256 + n] = fmaxf(acc[r] + bias, 0.0f);
        }
    }
}

// ---------------------------------------------------------------------------
extern "C" void kernel_launch(void* const* d_in, const int* in_sizes, int n_in,
                              void* d_out, int out_size, void* d_ws, size_t ws_size,
                              hipStream_t stream) {
    const float* state = (const float*)d_in[0];   // [512,512]
    const float* img   = (const float*)d_in[1];   // [512,224,224,1]
    const float* W_loc = (const float*)d_in[2];
    const float* b_loc = (const float*)d_in[3];
    const float* W_hg  = (const float*)d_in[4];
    const float* b_hg  = (const float*)d_in[5];
    const float* W_hl  = (const float*)d_in[6];
    const float* b_hl  = (const float*)d_in[7];
    const float* W_gs  = (const float*)d_in[8];
    const float* b_gs  = (const float*)d_in[9];
    const float* W_ls  = (const float*)d_in[10];
    const float* b_ls  = (const float*)d_in[11];
    float* out = (float*)d_out;

    // workspace layout (base is 256B aligned; pack regions are 32B aligned)
    char* ws = (char*)d_ws;
    float*     locf    = (float*)(ws + 0);                       // 4096 B
    int*       loci    = (int*)  (ws + 4096);                    // 4096 B
    _Float16*  packWhg = (_Float16*)(ws + 8192);                 // 196608 B
    _Float16*  packWc  = (_Float16*)(ws + 8192 + (size_t)N1PACK * 2); // 131072 B

    // pack weights (f16, fragment order)
    int packTot = N1PACK + N2PACK;
    attn_pack_weights<<<(packTot + 255) / 256, 256, 0, stream>>>(
        W_hg, W_gs, W_ls, packWhg, packWc);

    // locations: 512 rows, one wave32 each -> 64 blocks of 256
    attn_loc<<<64, 256, 0, stream>>>(state, W_loc, b_loc, locf, loci);

    // fused glimpse + WMMA GEMMs: 512/16 = 32 blocks
    attn_main<<<32, 256, 0, stream>>>(img, W_hl, b_hl, b_hg, b_gs, b_ls,
                                      locf, loci, packWhg, packWc, out);
}
